// SSMLayer_62491774156947
// MI455X (gfx1250) — matrix operand order repaired and checked by reading, not compiled, not run on hardware
//
#include <hip/hip_runtime.h>
#include <math.h>

#define D_MODEL 1024
#define D_STATE 256
#define BATCH   8
#define SEQ     4096
#define M_ROWS  (BATCH*SEQ)     // 32768 token rows

#define CHUNK   256
#define NCHUNK  (SEQ/CHUNK)     // 16

typedef unsigned short u16;
typedef __attribute__((ext_vector_type(8)))  float   v8f;
typedef __attribute__((ext_vector_type(8)))  __bf16  v8bf;
typedef __attribute__((ext_vector_type(16))) __bf16  v16bf;

#define KC     64               // K-chunk staged in LDS per stage
#define PITCH  72               // LDS row pitch in bf16 units (144B, 16B-aligned rows)
#define PLANE  (128*PITCH)      // one bf16 plane (hi or lo) for a 128-row tile

// split f32 -> hi/lo bf16 (truncation; lo captures next 8 mantissa bits)
__device__ __forceinline__ void bsplit(float v, u16& hi, u16& lo) {
  unsigned u = __float_as_uint(v);
  hi = (u16)(u >> 16);
  float hf = __uint_as_float(u & 0xFFFF0000u);
  lo = (u16)(__float_as_uint(v - hf) >> 16);
}

__device__ __forceinline__ void stage_f4(u16* lds, int off, float4 v) {
  u16 h0,h1,h2,h3,l0,l1,l2,l3;
  bsplit(v.x,h0,l0); bsplit(v.y,h1,l1); bsplit(v.z,h2,l2); bsplit(v.w,h3,l3);
  uint2 hp, lp;
  hp.x = (unsigned)h0 | ((unsigned)h1 << 16);
  hp.y = (unsigned)h2 | ((unsigned)h3 << 16);
  lp.x = (unsigned)l0 | ((unsigned)l1 << 16);
  lp.y = (unsigned)l2 | ((unsigned)l3 << 16);
  *(uint2*)(&lds[off])         = hp;   // hi plane
  *(uint2*)(&lds[PLANE + off]) = lp;   // lo plane
}

__device__ __forceinline__ v16bf ldfrag(const u16* p0, const u16* p1) {
  v8bf a = *(const v8bf*)p0;
  v8bf b = *(const v8bf*)p1;
  return __builtin_shufflevector(a, b, 0,1,2,3,4,5,6,7,8,9,10,11,12,13,14,15);
}

#define WMMA_BF16(A, B, C) \
  __builtin_amdgcn_wmma_f32_16x16x32_bf16(false, (A), false, (B), (short)0, (C), false, false)

// ---------------------------------------------------------------------------
// GEMM1: Bx[m,n] = sum_d x[m,d] * B[n,d]   (M=32768, N=256, K=1024)
// block = 256 thr (8 waves); block tile 128m x 128n; wave: 16m x 128n.
// f32 emulated as bf16 hi/lo split: 3 bf16 WMMAs per 16x16x32.
// ---------------------------------------------------------------------------
__global__ __launch_bounds__(256)
void ssm_gemm_bx(const float* __restrict__ x,
                 const float* __restrict__ Bmat,
                 float* __restrict__ Bx) {
  __shared__ __align__(16) u16 lsA[2*PLANE];   // x tile   128 x 64, hi+lo planes
  __shared__ __align__(16) u16 lsB[2*PLANE];   // B tile   128 x 64 (row=n, col=k)

  const int tid  = threadIdx.x;
  const int wave = tid >> 5;
  const int lane = tid & 31;
  const int half = lane >> 4;
  const int l16  = lane & 15;
  const int m0   = blockIdx.x * 128;
  const int n0   = blockIdx.y * 128;

  v8f acc[8];
  #pragma unroll
  for (int nt = 0; nt < 8; ++nt) {
    v8f z = {0.f,0.f,0.f,0.f,0.f,0.f,0.f,0.f};
    acc[nt] = z;
  }

  for (int kc = 0; kc < D_MODEL; kc += KC) {
    // stage x tile: 128 rows x 64 k = 2048 float4, 8 per thread
    #pragma unroll
    for (int i = 0; i < 8; ++i) {
      int f   = tid + i * 256;
      int row = f >> 4;          // 16 float4 per row
      int c4  = f & 15;
      float4 v = *(const float4*)(x + (size_t)(m0 + row) * D_MODEL + kc + c4 * 4);
      stage_f4(lsA, row * PITCH + c4 * 4, v);
    }
    // stage B tile: 128 rows x 64 k
    #pragma unroll
    for (int i = 0; i < 8; ++i) {
      int f   = tid + i * 256;
      int row = f >> 4;
      int c4  = f & 15;
      float4 v = *(const float4*)(Bmat + (size_t)(n0 + row) * D_MODEL + kc + c4 * 4);
      stage_f4(lsB, row * PITCH + c4 * 4, v);
    }
    __syncthreads();

    #pragma unroll
    for (int kt = 0; kt < 2; ++kt) {
      const int ko = kt * 32;
      // A frag (bf16 16x32): lane<16 -> K 0-7 & 16-23 ; lane>=16 -> K 8-15 & 24-31
      const u16* ap = &lsA[(wave * 16 + l16) * PITCH + ko + half * 8];
      v16bf ahi = ldfrag(ap,         ap + 16);
      v16bf alo = ldfrag(ap + PLANE, ap + PLANE + 16);
      #pragma unroll
      for (int nt = 0; nt < 8; ++nt) {
        // B frag (bf16 32x16): lane<16 -> K 0-7 & 16-23 ; lane>=16 -> K 8-15 & 24-31
        const u16* bp = &lsB[(nt * 16 + l16) * PITCH + ko + half * 8];
        v16bf bhi = ldfrag(bp,         bp + 16);
        v16bf blo = ldfrag(bp + PLANE, bp + PLANE + 16);
        acc[nt] = WMMA_BF16(ahi, bhi, acc[nt]);
        acc[nt] = WMMA_BF16(alo, bhi, acc[nt]);
        acc[nt] = WMMA_BF16(ahi, blo, acc[nt]);
      }
    }
    __syncthreads();
  }

  // store: VGPR r -> M = r (+8 for upper lanes), N = l16
  #pragma unroll
  for (int nt = 0; nt < 8; ++nt) {
    int col   = n0 + nt * 16 + l16;
    int rbase = m0 + wave * 16 + half * 8;
    #pragma unroll
    for (int r = 0; r < 8; ++r) {
      Bx[(size_t)(rbase + r) * D_STATE + col] = acc[nt][r];
    }
  }
}

// ---------------------------------------------------------------------------
// Scan, chunk-parallel: seq split into 16 chunks of 256.
// Phase A: chunk-local scan (h0=0), in place; emit chunk-final h.
// Phase B: combine carries per (b,n): carry_in[c] = a^256 * carry_in[c-1] + fin[c-1]
// Phase C (power table): P[tloc][n] = a[n]^(tloc+1)
// The fixup hs += P[tloc]*carry_in is fused into GEMM2's staging (saves a
// 67 MB global RMW pass; P and carries are L2-resident).
// ---------------------------------------------------------------------------
__global__ __launch_bounds__(256)
void ssm_scan_local(float* __restrict__ buf, float* __restrict__ carries,
                    const float* __restrict__ A) {
  const int n = threadIdx.x;
  const int b = blockIdx.x >> 4;
  const int c = blockIdx.x & (NCHUNK - 1);
  const float a = expf(-expf(A[n]));
  float h = 0.f;
  size_t base = ((size_t)(b * SEQ + c * CHUNK)) * D_STATE + n;
  for (int t = 0; t < CHUNK; t += 4) {
    float v0 = buf[base + (size_t)(t + 0) * D_STATE];
    float v1 = buf[base + (size_t)(t + 1) * D_STATE];
    float v2 = buf[base + (size_t)(t + 2) * D_STATE];
    float v3 = buf[base + (size_t)(t + 3) * D_STATE];
    h = fmaf(a, h, v0); buf[base + (size_t)(t + 0) * D_STATE] = h;
    h = fmaf(a, h, v1); buf[base + (size_t)(t + 1) * D_STATE] = h;
    h = fmaf(a, h, v2); buf[base + (size_t)(t + 2) * D_STATE] = h;
    h = fmaf(a, h, v3); buf[base + (size_t)(t + 3) * D_STATE] = h;
  }
  carries[(size_t)(b * NCHUNK + c) * D_STATE + n] = h;
}

__global__ __launch_bounds__(256)
void ssm_scan_carry(float* __restrict__ carries, const float* __restrict__ A) {
  const int n = threadIdx.x;
  const int b = blockIdx.x;
  const float a = expf(-expf(A[n]));
  float aC = a;
  #pragma unroll
  for (int i = 0; i < 8; ++i) aC *= aC;    // a^256
  float p = 0.f;
  for (int c = 0; c < NCHUNK; ++c) {
    size_t idx = (size_t)(b * NCHUNK + c) * D_STATE + n;
    float fin = carries[idx];
    carries[idx] = p;                       // becomes carry_in for chunk c
    p = fmaf(aC, p, fin);
  }
}

__global__ __launch_bounds__(256)
void ssm_pow(float* __restrict__ P, const float* __restrict__ A) {
  const int n = threadIdx.x;
  const float a = expf(-expf(A[n]));
  float g = a;                              // a^(tloc+1), tloc = 0
  for (int t = 0; t < CHUNK; ++t) {
    P[(size_t)t * D_STATE + n] = g;
    g *= a;
  }
}

// ---------------------------------------------------------------------------
// GEMM2: y[m,d] = sum_n hs_true[m,n] * C[d,n] + D[d] * x[m,d]
// (M=32768, N=1024, K=256). hs_true = hs_local + P[tloc]*carry_in, applied
// on the fly while staging (before the bf16 hi/lo split).
// ---------------------------------------------------------------------------
__global__ __launch_bounds__(256)
void ssm_gemm_y(const float* __restrict__ hs,
                const float* __restrict__ carries,
                const float* __restrict__ Ppow,
                const float* __restrict__ Cmat,
                const float* __restrict__ x,
                const float* __restrict__ Dvec,
                float* __restrict__ y) {
  __shared__ __align__(16) u16 lsA[2*PLANE];   // hs tile 128 x 64
  __shared__ __align__(16) u16 lsB[2*PLANE];   // C tile  128 x 64 (row=d, col=n)

  const int tid  = threadIdx.x;
  const int wave = tid >> 5;
  const int lane = tid & 31;
  const int half = lane >> 4;
  const int l16  = lane & 15;
  const int m0   = blockIdx.x * 128;
  const int d0   = blockIdx.y * 128;

  v8f acc[8];
  #pragma unroll
  for (int nt = 0; nt < 8; ++nt) {
    v8f z = {0.f,0.f,0.f,0.f,0.f,0.f,0.f,0.f};
    acc[nt] = z;
  }

  for (int kc = 0; kc < D_STATE; kc += KC) {
    #pragma unroll
    for (int i = 0; i < 8; ++i) {
      int f    = tid + i * 256;
      int row  = f >> 4;
      int c4   = f & 15;
      int grow = m0 + row;            // global token row
      int tloc = grow & (CHUNK - 1);  // position within chunk
      int gch  = grow >> 8;           // global chunk id = b*NCHUNK + c
      int nn   = kc + c4 * 4;         // state-dim column
      float4 v  = *(const float4*)(hs      + (size_t)grow * D_STATE + nn);
      float4 p  = *(const float4*)(Ppow    + (size_t)tloc * D_STATE + nn);
      float4 ci = *(const float4*)(carries + (size_t)gch  * D_STATE + nn);
      v.x = fmaf(p.x, ci.x, v.x);
      v.y = fmaf(p.y, ci.y, v.y);
      v.z = fmaf(p.z, ci.z, v.z);
      v.w = fmaf(p.w, ci.w, v.w);
      stage_f4(lsA, row * PITCH + c4 * 4, v);
    }
    #pragma unroll
    for (int i = 0; i < 8; ++i) {
      int f   = tid + i * 256;
      int row = f >> 4;
      int c4  = f & 15;
      float4 v = *(const float4*)(Cmat + (size_t)(d0 + row) * D_STATE + kc + c4 * 4);
      stage_f4(lsB, row * PITCH + c4 * 4, v);
    }
    __syncthreads();

    #pragma unroll
    for (int kt = 0; kt < 2; ++kt) {
      const int ko = kt * 32;
      const u16* ap = &lsA[(wave * 16 + l16) * PITCH + ko + half * 8];
      v16bf ahi = ldfrag(ap,         ap + 16);
      v16bf alo = ldfrag(ap + PLANE, ap + PLANE + 16);
      #pragma unroll
      for (int nt = 0; nt < 8; ++nt) {
        const u16* bp = &lsB[(nt * 16 + l16) * PITCH + ko + half * 8];
        v16bf bhi = ldfrag(bp,         bp + 16);
        v16bf blo = ldfrag(bp + PLANE, bp + PLANE + 16);
        acc[nt] = WMMA_BF16(ahi, bhi, acc[nt]);
        acc[nt] = WMMA_BF16(alo, bhi, acc[nt]);
        acc[nt] = WMMA_BF16(ahi, blo, acc[nt]);
      }
    }
    __syncthreads();
  }

  // epilogue: y = acc + D[d] * x[m,d]
  #pragma unroll
  for (int nt = 0; nt < 8; ++nt) {
    int col   = d0 + nt * 16 + l16;
    int rbase = m0 + wave * 16 + half * 8;
    float dv  = Dvec[col];
    #pragma unroll
    for (int r = 0; r < 8; ++r) {
      size_t idx = (size_t)(rbase + r) * D_MODEL + col;
      y[idx] = acc[nt][r] + dv * x[idx];
    }
  }
}

// ---------------------------------------------------------------------------
extern "C" void kernel_launch(void* const* d_in, const int* in_sizes, int n_in,
                              void* d_out, int out_size, void* d_ws, size_t ws_size,
                              hipStream_t stream) {
  const float* x    = (const float*)d_in[0];
  const float* A    = (const float*)d_in[1];
  const float* Bmat = (const float*)d_in[2];
  const float* Cmat = (const float*)d_in[3];
  const float* Dvec = (const float*)d_in[4];
  float*       y    = (float*)d_out;

  float* Bx      = (float*)d_ws;                              // 33.5 MB
  float* carries = Bx + (size_t)M_ROWS * D_STATE;             // 128 KB
  float* Ppow    = carries + (size_t)BATCH * NCHUNK * D_STATE; // 256 KB

  ssm_gemm_bx   <<<dim3(M_ROWS/128, D_STATE/128), 256, 0, stream>>>(x, Bmat, Bx);
  ssm_scan_local<<<dim3(BATCH*NCHUNK), 256, 0, stream>>>(Bx, carries, A);
  ssm_scan_carry<<<dim3(BATCH),        256, 0, stream>>>(carries, A);
  ssm_pow       <<<dim3(1),            256, 0, stream>>>(Ppow, A);
  ssm_gemm_y    <<<dim3(M_ROWS/128, D_MODEL/128), 256, 0, stream>>>(
      Bx, carries, Ppow, Cmat, x, Dvec, y);
}
